// MemoryAugmentedAttention_69879117906504
// MI455X (gfx1250) — matrix operand order repaired and checked by reading, not compiled, not run on hardware
//
#include <hip/hip_runtime.h>

typedef _Float16 half_t;
typedef __attribute__((ext_vector_type(16))) _Float16 v16h;
typedef __attribute__((ext_vector_type(8)))  _Float16 v8h;
typedef __attribute__((ext_vector_type(4)))  _Float16 v4h;
typedef __attribute__((ext_vector_type(8)))  float    v8f;

typedef __attribute__((address_space(3))) half_t lds_half_t;

#define D_MODEL 1024
#define NHEAD   16
#define DK      64
#define BATCH   2
#define SEQ     2048   // 2*S after concat
#define S_HALF  1024

union V16U { v16h v; v8h h[2]; };

// ---------------------------------------------------------------------------
// WMMA helper: D = A(16x32 f16) * B(32x16 f16) + C(16x16 f32)
// ---------------------------------------------------------------------------
__device__ __forceinline__ v8f wmma_f32_f16(v16h a, v16h b, v8f c) {
  return __builtin_amdgcn_wmma_f32_16x16x32_f16(
      /*neg_a=*/false, a, /*neg_b=*/false, b,
      /*c_mod=*/(short)0, c, /*reuse_a=*/false, /*reuse_b=*/false);
}

// A-matrix fragment (16 rows x 32 k), row-major source with leading dim ld.
// Lanes 0-15 row M=lane hold K{0..7,16..23}; lanes 16-31 hold K{8..15,24..31}.
__device__ __forceinline__ v16h ld_frag_a(const half_t* base, int ld, int lane) {
  int r  = lane & 15;
  int ko = (lane < 16) ? 0 : 8;
  V16U u;
  u.h[0] = *(const v8h*)(base + r * ld + ko);
  u.h[1] = *(const v8h*)(base + r * ld + ko + 16);
  return u.v;
}

// B-matrix fragment (32 k x 16 cols). Source holds the B^T view row-major.
// Lanes 0-15 col N=lane hold K{0..7,16..23}; lanes 16-31 hold K{8..15,24..31}.
__device__ __forceinline__ v16h ld_frag_b(const half_t* base, int ld, int lane) {
  int r  = lane & 15;
  int ko = (lane < 16) ? 0 : 8;
  V16U u;
  u.h[0] = *(const v8h*)(base + r * ld + ko);
  u.h[1] = *(const v8h*)(base + r * ld + ko + 16);
  return u.v;
}

// ---------------------------------------------------------------------------
// CDNA5 async memory -> LDS copy (ASYNCcnt-tracked, 16B per lane)
// GLOBAL_LOAD_ASYNC_TO_LDS_B128, GV mode (SADDR = off)
// ---------------------------------------------------------------------------
__device__ __forceinline__ unsigned lds_addr_of(half_t* p) {
  return (unsigned)(unsigned long long)(lds_half_t*)p;
}
__device__ __forceinline__ void async_copy_b128(half_t* lds_dst,
                                                const half_t* g_src) {
  unsigned l = lds_addr_of(lds_dst);
  asm volatile("global_load_async_to_lds_b128 %0, %1, off"
               : : "v"(l), "v"(g_src) : "memory");
}
__device__ __forceinline__ void wait_async_zero() {
  asm volatile("s_wait_asynccnt 0x0" ::: "memory");
}

// ---------------------------------------------------------------------------
// f32 -> f16 conversion (grid-stride, vectorized)
// ---------------------------------------------------------------------------
__global__ void cvt_f32_f16(const float* __restrict__ src,
                            half_t* __restrict__ dst, int n4) {
  int i = blockIdx.x * blockDim.x + threadIdx.x;
  int stride = gridDim.x * blockDim.x;
  for (; i < n4; i += stride) {
    float4 v = ((const float4*)src)[i];
    v4h o;
    o[0] = (half_t)v.x; o[1] = (half_t)v.y;
    o[2] = (half_t)v.z; o[3] = (half_t)v.w;
    ((v4h*)dst)[i] = o;
  }
}

// ---------------------------------------------------------------------------
// Tiled WMMA GEMM: out[M,N] = act(A[M,K] @ W[N,K]^T + bias[N])
// Block: 256 threads = 8 waves (2x4), 64x128 block tile, wave tile 32x32.
// REMAP: A row = out_row + 1024 + ((out_row>>10)<<10)  (out[:, -S:, :] slice)
// ---------------------------------------------------------------------------
template <bool RELU, bool OUTF32, bool REMAP>
__global__ __launch_bounds__(256) void
gemm_wmma(const half_t* __restrict__ A,
          const half_t* __restrict__ W,
          const float* __restrict__ bias,
          half_t* __restrict__ outh,
          float* __restrict__ outf,
          int M, int N, int K) {
  __shared__ half_t As[64][32];
  __shared__ half_t Ws[128][32];

  const int tid  = threadIdx.x;
  const int wave = tid >> 5;
  const int lane = tid & 31;
  const int wm = wave & 1;      // 0..1 : 32-row strip
  const int wn = wave >> 1;     // 0..3 : 32-col strip
  const int bm = blockIdx.x * 64;
  const int bn = blockIdx.y * 128;

  v8f acc[2][2] = {};

  for (int k0 = 0; k0 < K; k0 += 32) {
    __syncthreads();
    // Stage A tile 64x32 via async DMA to LDS (8 halves = 16B per thread)
    {
      int idx = tid * 8;
      int r = idx >> 5, c = idx & 31;
      int gr = bm + r;
      int arow = REMAP ? (gr + 1024 + ((gr >> 10) << 10)) : gr;
      async_copy_b128(&As[r][c], A + (size_t)arow * K + k0 + c);
      if (k0 + 32 < K)
        __builtin_prefetch(A + (size_t)arow * K + k0 + 32 + c, 0, 0);
    }
    // Stage W tile 128x32 (2 x 16B per thread)
    for (int i = 0; i < 2; ++i) {
      int idx = tid * 16 + i * 8;
      int r = idx >> 5, c = idx & 31;
      async_copy_b128(&Ws[r][c], W + (size_t)(bn + r) * K + k0 + c);
    }
    wait_async_zero();
    __syncthreads();

    v16h a0 = ld_frag_a(&As[wm * 32][0], 32, lane);
    v16h a1 = ld_frag_a(&As[wm * 32 + 16][0], 32, lane);
    v16h b0 = ld_frag_b(&Ws[wn * 32][0], 32, lane);
    v16h b1 = ld_frag_b(&Ws[wn * 32 + 16][0], 32, lane);

    acc[0][0] = wmma_f32_f16(a0, b0, acc[0][0]);
    acc[0][1] = wmma_f32_f16(a0, b1, acc[0][1]);
    acc[1][0] = wmma_f32_f16(a1, b0, acc[1][0]);
    acc[1][1] = wmma_f32_f16(a1, b1, acc[1][1]);
  }

  // Epilogue: bias + activation + store
  const int n0 = bn + wn * 32;
  const float bias0 = bias[n0 + (lane & 15)];
  const float bias1 = bias[n0 + 16 + (lane & 15)];
  const int half_ofs = (lane < 16) ? 0 : 8;
#pragma unroll
  for (int tm = 0; tm < 2; ++tm)
#pragma unroll
    for (int tn = 0; tn < 2; ++tn)
#pragma unroll
      for (int r = 0; r < 8; ++r) {
        int m = bm + wm * 32 + tm * 16 + r + half_ofs;
        int n = n0 + tn * 16 + (lane & 15);
        float v = acc[tm][tn][r] + (tn ? bias1 : bias0);
        if (RELU) v = fmaxf(v, 0.0f);
        if (OUTF32) outf[(size_t)m * N + n] = v;
        else        outh[(size_t)m * N + n] = (half_t)v;
      }
}

// ---------------------------------------------------------------------------
// Attention: per block = (q-block of 64 rows, head, batch), 4 waves x 16 rows.
// Pass 1: flash-style online softmax, ctx = softmax(QK^T/sqrt(dk)) @ V (f16).
// Pass 2: recompute score tiles, stream normalized probabilities to attn out
//         with non-temporal stores (536MB write-once >> 192MB L2).
// Q fragments are pre-scaled by 1/sqrt(64)=0.125 (exact power of 2 in f16).
// ---------------------------------------------------------------------------
__device__ __forceinline__ float red16_max(float v) {
  v = fmaxf(v, __shfl_xor(v, 1, 32));
  v = fmaxf(v, __shfl_xor(v, 2, 32));
  v = fmaxf(v, __shfl_xor(v, 4, 32));
  v = fmaxf(v, __shfl_xor(v, 8, 32));
  return v;
}
__device__ __forceinline__ float red16_sum(float v) {
  v += __shfl_xor(v, 1, 32);
  v += __shfl_xor(v, 2, 32);
  v += __shfl_xor(v, 4, 32);
  v += __shfl_xor(v, 8, 32);
  return v;
}

__global__ __launch_bounds__(128, 1) void
attn_wmma(const half_t* __restrict__ Q,
          const half_t* __restrict__ Km,
          const half_t* __restrict__ V,
          float* __restrict__ attn_out,
          half_t* __restrict__ ctx_out) {
  __shared__ half_t Kt[64][64];       // keys x dk (row-major, K-contiguous)
  __shared__ half_t VT[64][64];       // dk x keys (transposed V tile)
  __shared__ half_t Pl[4][16][64];    // per-wave P repack buffer

  const int tid  = threadIdx.x;
  const int wave = tid >> 5;
  const int lane = tid & 31;
  const int qb = blockIdx.x;          // 0..31
  const int h  = blockIdx.y;          // 0..15
  const int b  = blockIdx.z;          // 0..1
  const size_t rowbase = (size_t)b * SEQ;
  const int hcol = h * DK;
  const int qrow0 = qb * 64 + wave * 16;
  const int half_ofs = (lane < 16) ? 0 : 8;

  // Q fragments for the wave's 16 rows (dk 0-31, 32-63), loaded once and
  // pre-scaled by 1/sqrt(dk) = 0.125 (exact in f16).
  v16h qa[2];
  {
    const half_t* qptr = Q + (rowbase + qrow0 + (lane & 15)) * D_MODEL + hcol;
#pragma unroll
    for (int kc = 0; kc < 2; ++kc) {
      int ko = ((lane < 16) ? 0 : 8) + kc * 32;
      V16U u;
      u.h[0] = *(const v8h*)(qptr + ko);
      u.h[1] = *(const v8h*)(qptr + ko + 16);
      qa[kc] = u.v * (half_t)0.125f;
    }
  }

  float mrow[8], lrow[8];
#pragma unroll
  for (int r = 0; r < 8; ++r) { mrow[r] = -1e30f; lrow[r] = 0.0f; }
  v8f ctx[4] = {};

  // ------------------------------ pass 1 ------------------------------
  for (int ks = 0; ks < SEQ; ks += 64) {
    __syncthreads();
    // K tile (64 keys x 64 dk): async DMA straight into LDS
#pragma unroll
    for (int i = 0; i < 4; ++i) {
      int idx = tid * 4 + i;
      int rr = idx >> 3, cc = (idx & 7) * 8;
      async_copy_b128(&Kt[rr][cc],
                      Km + (rowbase + ks + rr) * D_MODEL + hcol + cc);
    }
    // V tile transposed while the K DMA is in flight:
    // thread t handles key pair (2*kp, 2*kp+1) x 16 dk values -> packed b32
    {
      int kp = tid & 31;
      int d0 = (tid >> 5) * 16;
      const half_t* v0 = V + (rowbase + ks + 2 * kp) * D_MODEL + hcol + d0;
      const half_t* v1 = v0 + D_MODEL;
      half_t ta[16], tb[16];
      *(v8h*)&ta[0] = *(const v8h*)(v0);
      *(v8h*)&ta[8] = *(const v8h*)(v0 + 8);
      *(v8h*)&tb[0] = *(const v8h*)(v1);
      *(v8h*)&tb[8] = *(const v8h*)(v1 + 8);
#pragma unroll
      for (int i = 0; i < 16; ++i) {
        union { half_t h[2]; unsigned u; } pk;
        pk.h[0] = ta[i]; pk.h[1] = tb[i];
        *(unsigned*)&VT[d0 + i][2 * kp] = pk.u;
      }
    }
    wait_async_zero();
    __syncthreads();

    // Scores: S(16x64) = (Q*scale)(16x64) @ K^T
    v8f s[4];
#pragma unroll
    for (int kt = 0; kt < 4; ++kt) {
      v8f a = {};
#pragma unroll
      for (int kc = 0; kc < 2; ++kc) {
        v16h bf = ld_frag_b(&Kt[kt * 16][kc * 32], 64, lane);
        a = wmma_f32_f16(qa[kc], bf, a);
      }
      s[kt] = a;
    }

    // Online softmax update (row r lives in VGPR r of each C tile; the
    // 16-lane row groups match the WMMA C layout exactly)
    float alpha[8];
#pragma unroll
    for (int r = 0; r < 8; ++r) {
      float tm = fmaxf(fmaxf(s[0][r], s[1][r]), fmaxf(s[2][r], s[3][r]));
      tm = red16_max(tm);
      float mn = fmaxf(mrow[r], tm);
      alpha[r] = __expf(mrow[r] - mn);
      mrow[r] = mn;
    }
#pragma unroll
    for (int kt = 0; kt < 4; ++kt)
#pragma unroll
      for (int r = 0; r < 8; ++r)
        s[kt][r] = __expf(s[kt][r] - mrow[r]);
#pragma unroll
    for (int r = 0; r < 8; ++r) {
      float sum = s[0][r] + s[1][r] + s[2][r] + s[3][r];
      sum = red16_sum(sum);
      lrow[r] = lrow[r] * alpha[r] + sum;
    }
#pragma unroll
    for (int nt = 0; nt < 4; ++nt)
#pragma unroll
      for (int r = 0; r < 8; ++r)
        ctx[nt][r] *= alpha[r];

    // Repack P (C layout f32) -> A layout f16 via per-wave LDS
#pragma unroll
    for (int kt = 0; kt < 4; ++kt)
#pragma unroll
      for (int r = 0; r < 8; ++r)
        Pl[wave][r + half_ofs][kt * 16 + (lane & 15)] = (half_t)s[kt][r];
    asm volatile("s_wait_dscnt 0" ::: "memory");

    // ctx += P(16x64) @ V(64x64)
#pragma unroll
    for (int kc = 0; kc < 2; ++kc) {
      v16h pa = ld_frag_a(&Pl[wave][0][kc * 32], 64, lane);
#pragma unroll
      for (int nt = 0; nt < 4; ++nt) {
        v16h vb = ld_frag_b(&VT[nt * 16][kc * 32], 64, lane);
        ctx[nt] = wmma_f32_f16(pa, vb, ctx[nt]);
      }
    }
  }

  // Finalize ctx and store f16
  float inv[8];
#pragma unroll
  for (int r = 0; r < 8; ++r) inv[r] = 1.0f / lrow[r];
#pragma unroll
  for (int nt = 0; nt < 4; ++nt)
#pragma unroll
    for (int r = 0; r < 8; ++r) {
      int m = qrow0 + r + half_ofs;
      int n = hcol + nt * 16 + (lane & 15);
      ctx_out[(rowbase + m) * D_MODEL + n] = (half_t)(ctx[nt][r] * inv[r]);
    }

  // ------------------------------ pass 2 ------------------------------
  // attn[b][h][q][k] = exp(s - m) / l  (recompute scores, stream to HBM)
  float* aout = attn_out + ((size_t)(b * NHEAD + h) * SEQ) * SEQ;
  for (int ks = 0; ks < SEQ; ks += 64) {
    __syncthreads();
#pragma unroll
    for (int i = 0; i < 4; ++i) {
      int idx = tid * 4 + i;
      int rr = idx >> 3, cc = (idx & 7) * 8;
      async_copy_b128(&Kt[rr][cc],
                      Km + (rowbase + ks + rr) * D_MODEL + hcol + cc);
    }
    wait_async_zero();
    __syncthreads();
#pragma unroll
    for (int kt = 0; kt < 4; ++kt) {
      v8f a = {};
#pragma unroll
      for (int kc = 0; kc < 2; ++kc) {
        v16h bf = ld_frag_b(&Kt[kt * 16][kc * 32], 64, lane);
        a = wmma_f32_f16(qa[kc], bf, a);
      }
#pragma unroll
      for (int r = 0; r < 8; ++r) {
        float p = __expf(a[r] - mrow[r]) * inv[r];
        int q = qrow0 + r + half_ofs;
        int k = ks + kt * 16 + (lane & 15);
        __builtin_nontemporal_store(p, aout + (size_t)q * SEQ + k);
      }
    }
  }
}

// ---------------------------------------------------------------------------
// Host-side orchestration
// ---------------------------------------------------------------------------
extern "C" void kernel_launch(void* const* d_in, const int* in_sizes, int n_in,
                              void* d_out, int out_size, void* d_ws, size_t ws_size,
                              hipStream_t stream) {
  (void)in_sizes; (void)n_in; (void)out_size; (void)ws_size;
  const float* x   = (const float*)d_in[0];
  const float* Wq  = (const float*)d_in[1];
  const float* bq  = (const float*)d_in[2];
  const float* Wk  = (const float*)d_in[3];
  const float* bk  = (const float*)d_in[4];
  const float* Wv  = (const float*)d_in[5];
  const float* bv  = (const float*)d_in[6];
  const float* Wo  = (const float*)d_in[7];
  const float* bo  = (const float*)d_in[8];
  const float* M1  = (const float*)d_in[9];
  const float* bm1 = (const float*)d_in[10];
  const float* M2  = (const float*)d_in[11];
  const float* bm2 = (const float*)d_in[12];

  const size_t MM = (size_t)1024 * 1024;  // 1M elements (weight matrices)
  half_t* ws   = (half_t*)d_ws;
  half_t* enh  = ws;                 // 4096x1024 (rows 0-2047 mem, 2048-4095 x)
  half_t* hbuf = enh  + 4 * MM;      // 2048x1024
  half_t* wq_h = hbuf + 2 * MM;
  half_t* wk_h = wq_h + MM;
  half_t* wv_h = wk_h + MM;
  half_t* wo_h = wv_h + MM;
  half_t* m1_h = wo_h + MM;
  half_t* m2_h = m1_h + MM;
  half_t* Qb   = m2_h + MM;          // 4096x1024
  half_t* Kb   = Qb   + 4 * MM;
  half_t* Vb   = Kb   + 4 * MM;
  half_t* ctx  = Vb   + 4 * MM;      // 4096x1024

  float* out_f = (float*)d_out;                 // out_orig: 2048x1024
  float* attn  = out_f + 2 * MM;                // attn: 2x16x2048x2048

  const dim3 cvb(256), cvg(1024);
  hipLaunchKernelGGL(cvt_f32_f16, cvg, cvb, 0, stream, x,  enh + 2 * MM, (int)(2 * MM / 4));
  hipLaunchKernelGGL(cvt_f32_f16, cvg, cvb, 0, stream, Wq, wq_h, (int)(MM / 4));
  hipLaunchKernelGGL(cvt_f32_f16, cvg, cvb, 0, stream, Wk, wk_h, (int)(MM / 4));
  hipLaunchKernelGGL(cvt_f32_f16, cvg, cvb, 0, stream, Wv, wv_h, (int)(MM / 4));
  hipLaunchKernelGGL(cvt_f32_f16, cvg, cvb, 0, stream, Wo, wo_h, (int)(MM / 4));
  hipLaunchKernelGGL(cvt_f32_f16, cvg, cvb, 0, stream, M1, m1_h, (int)(MM / 4));
  hipLaunchKernelGGL(cvt_f32_f16, cvg, cvb, 0, stream, M2, m2_h, (int)(MM / 4));

  const dim3 gblk(256);
  const dim3 g2k(2048 / 64, 1024 / 128);   // M=2048 GEMMs
  const dim3 g4k(4096 / 64, 1024 / 128);   // M=4096 GEMMs

  // h = relu(x @ M1^T + bm1)
  hipLaunchKernelGGL((gemm_wmma<true, false, false>), g2k, gblk, 0, stream,
                     enh + 2 * MM, m1_h, bm1, hbuf, (float*)nullptr, 2048, 1024, 1024);
  // mem = h @ M2^T + bm2  -> enh rows 0-2047
  hipLaunchKernelGGL((gemm_wmma<false, false, false>), g2k, gblk, 0, stream,
                     hbuf, m2_h, bm2, enh, (float*)nullptr, 2048, 1024, 1024);
  // Q/K/V = enh @ W^T + b
  hipLaunchKernelGGL((gemm_wmma<false, false, false>), g4k, gblk, 0, stream,
                     enh, wq_h, bq, Qb, (float*)nullptr, 4096, 1024, 1024);
  hipLaunchKernelGGL((gemm_wmma<false, false, false>), g4k, gblk, 0, stream,
                     enh, wk_h, bk, Kb, (float*)nullptr, 4096, 1024, 1024);
  hipLaunchKernelGGL((gemm_wmma<false, false, false>), g4k, gblk, 0, stream,
                     enh, wv_h, bv, Vb, (float*)nullptr, 4096, 1024, 1024);

  // attention: ctx (f16) + attn (f32, non-temporal stream)
  hipLaunchKernelGGL(attn_wmma, dim3(SEQ / 64, NHEAD, BATCH), dim3(128), 0, stream,
                     Qb, Kb, Vb, attn, ctx);

  // out_orig = ctx[:, -S:, :] @ Wo^T + bo  (f32, row-remapped A)
  hipLaunchKernelGGL((gemm_wmma<false, true, true>), g2k, gblk, 0, stream,
                     ctx, wo_h, bo, (half_t*)nullptr, out_f, 2048, 1024, 1024);
}
